// SIGMA_46059229283037
// MI455X (gfx1250) — matrix operand order repaired and checked by reading, not compiled, not run on hardware
//
#include <hip/hip_runtime.h>
#include <hip/hip_bf16.h>
#include <math.h>

// ---------------- constants (match reference problem sizes) ----------------
#define N1C   1000      // nodes per graph
#define DIMC  256       // feature dim
#define NEC   16000     // edges per graph
#define NSC   8         // gumbel-sinkhorn samples
#define PDC   2000      // padded sinkhorn dim (n1+n2)
#define TITER 2
#define SINK_ITERS 10
#define TAUF  0.1f
#define EPSF  1e-20f
#define MISSV (-1.0f)
#define SIMSCALE 50.0f
#define BS 256

// ---------------- CDNA5 WMMA types ----------------
typedef __attribute__((ext_vector_type(16))) __bf16 v16bf;
typedef __attribute__((ext_vector_type(8)))  float  v8f;

__device__ __forceinline__ unsigned short f2bf(float f) {
  unsigned int u = __float_as_uint(f);
  unsigned int r = (u + 0x7FFFu + ((u >> 16) & 1u)) >> 16;  // round-to-nearest-even
  return (unsigned short)r;
}

__device__ __forceinline__ unsigned int hash3(unsigned int a, unsigned int b, unsigned int c) {
  a *= 0xcc9e2d51u; a = (a << 15) | (a >> 17); a *= 0x1b873593u;
  b ^= a; b = b * 5u + 0xe6546b64u;
  c ^= b; c *= 0x85ebca6bu; c ^= c >> 13; c *= 0xc2b2ae35u; c ^= c >> 16;
  return c;
}

// ---------------- generic fill ----------------
__global__ void k_fill(float* __restrict__ p, long long n, float v) {
  long long i = (long long)blockIdx.x * blockDim.x + threadIdx.x;
  if (i < n) p[i] = v;
}

// ---------------- h = x * importance (row scale) ----------------
__global__ void k_scale_rows(const float* __restrict__ x, const float* __restrict__ imp,
                             float* __restrict__ h, int d) {
  int r = blockIdx.x;
  float s = imp[r];
  for (int k = threadIdx.x; k < d; k += blockDim.x)
    h[(size_t)r * d + k] = x[(size_t)r * d + k] * s;
}

// ---------------- segment_sum over edges: agg[dst] += h[src] ----------------
__global__ void k_edge_agg(const float* __restrict__ h, const int* __restrict__ src,
                           const int* __restrict__ dst, float* __restrict__ agg, int d) {
  int e = blockIdx.x;
  int s = src[e], t = dst[e];
  const float* hs = h + (size_t)s * d;
  float* at = agg + (size_t)t * d;
  for (int k = threadIdx.x; k < d; k += blockDim.x)
    atomicAdd(&at[k], hs[k]);
}

// ---------------- row / col sums (importance for it>0) ----------------
__global__ void k_row_sum(const float* __restrict__ M, float* __restrict__ out, int C) {
  __shared__ float red[BS];
  int r = blockIdx.x;
  float s = 0.f;
  for (int j = threadIdx.x; j < C; j += blockDim.x) s += M[(size_t)r * C + j];
  red[threadIdx.x] = s; __syncthreads();
  for (int k = BS / 2; k > 0; k >>= 1) {
    if ((int)threadIdx.x < k) red[threadIdx.x] += red[threadIdx.x + k];
    __syncthreads();
  }
  if (threadIdx.x == 0) out[r] = red[0];
}

__global__ void k_col_sum(const float* __restrict__ M, float* __restrict__ out, int R, int C) {
  int c = blockIdx.x * blockDim.x + threadIdx.x;
  if (c >= C) return;
  float s = 0.f;
  for (int i = 0; i < R; i++) s += M[(size_t)i * C + c];
  out[c] = s;
}

// ---------------- L2 row normalize ----------------
__global__ void k_rownorm(float* __restrict__ h, int d) {
  __shared__ float red[BS];
  int r = blockIdx.x;
  float s = 0.f;
  for (int k = threadIdx.x; k < d; k += blockDim.x) {
    float v = h[(size_t)r * d + k];
    s += v * v;
  }
  red[threadIdx.x] = s; __syncthreads();
  for (int k = BS / 2; k > 0; k >>= 1) {
    if ((int)threadIdx.x < k) red[threadIdx.x] += red[threadIdx.x + k];
    __syncthreads();
  }
  float inv = rsqrtf(red[0]);
  for (int k = threadIdx.x; k < d; k += blockDim.x) h[(size_t)r * d + k] *= inv;
}

// ---------------- WMMA bf16 GEMM:  C = relu?( alpha*((A+A2) @ B[t]) + bias ) ----------------
// Block tile: 64x32 (M x N), K-step 64 (two 16x16x32 WMMAs per stage).
// 256 threads = 8 waves (wave32), wave grid 4x2, one 16x16 C tile per wave.
// LDS panels are staged in *fragment order* so each lane's 16-bf16 fragment is a
// single contiguous 32B chunk -> wide ds_load_b128 pairs instead of scalar gathers.
#define GTM 64
#define GTN 32
#define GTK 64
__global__ void k_gemm_wmma(const float* __restrict__ A, const float* __restrict__ A2, int lda,
                            const float* __restrict__ B, int ldb, int transB,
                            const float* __restrict__ bias, int relu, float alpha,
                            float* __restrict__ C, int ldc, int M, int N, int K) {
  // [m-tile][k-step][lane][elem] and [n-tile][k-step][lane][elem], bf16 bits
  __shared__ __align__(32) unsigned short As2[4][2][32][16];  // 8 KB
  __shared__ __align__(32) unsigned short Bs2[2][2][32][16];  // 4 KB

  const int m0 = blockIdx.y * GTM;
  const int n0 = blockIdx.x * GTN;
  const int wave = threadIdx.x >> 5;      // 0..7  (wave32)
  const int lane = threadIdx.x & 31;
  const int wm = wave & 3;                // wave row in tile grid (4)
  const int wn = wave >> 2;               // wave col in tile grid (2)
  const int hf = lane >> 4;               // lane half (0/1)
  const int lm = lane & 15;

  v8f acc = {};

  for (int kt = 0; kt < K; kt += GTK) {
    // ---- stage A panel (64 x 64) with fused A2 add, fragment-ordered ----
    for (int t = threadIdx.x; t < GTM * GTK; t += BS) {
      int m = t >> 6, k = t & 63;                 // coalesced along k
      int gm = m0 + m, gk = kt + k;
      float v = 0.f;
      if (gm < M && gk < K) {
        v = A[(size_t)gm * lda + gk];
        if (A2) v += A2[(size_t)gm * lda + gk];
      }
      int ks = k >> 5, kk = k & 31;
      int lw = (m & 15) + (((kk >> 3) & 1) << 4); // lane within wave
      int el = (kk & 7) + ((kk >> 4) << 3);       // element within fragment
      As2[m >> 4][ks][lw][el] = f2bf(v);
    }
    // ---- stage B panel (64 x 32), fragment-ordered ----
    if (!transB) {
      for (int t = threadIdx.x; t < GTK * GTN; t += BS) {
        int k = t >> 5, n = t & 31;               // coalesced along n
        int gk = kt + k, gn = n0 + n;
        float v = (gk < K && gn < N) ? B[(size_t)gk * ldb + gn] : 0.f;
        int ks = k >> 5, kk = k & 31;
        Bs2[n >> 4][ks][(n & 15) + (((kk >> 3) & 1) << 4)][(kk & 7) + ((kk >> 4) << 3)] = f2bf(v);
      }
    } else {
      for (int t = threadIdx.x; t < GTK * GTN; t += BS) {
        int n = t >> 6, k = t & 63;               // coalesced along k
        int gn = n0 + n, gk = kt + k;
        float v = (gk < K && gn < N) ? B[(size_t)gn * ldb + gk] : 0.f;
        int ks = k >> 5, kk = k & 31;
        Bs2[n >> 4][ks][(n & 15) + (((kk >> 3) & 1) << 4)][(kk & 7) + ((kk >> 4) << 3)] = f2bf(v);
      }
    }
    __syncthreads();

    // speculative prefetch of the next A panel (global_prefetch_b8)
    if (kt + GTK < K) {
      int gm = m0 + (int)(threadIdx.x & 63);
      if (gm < M) __builtin_prefetch(&A[(size_t)gm * lda + kt + GTK], 0, 1);
    }

    // ---- wide fragment loads (2x ds_load_b128 each) + 2 WMMAs ----
    v16bf fa0 = *(const v16bf*)&As2[wm][0][lane][0];
    v16bf fb0 = *(const v16bf*)&Bs2[wn][0][lane][0];
    v16bf fa1 = *(const v16bf*)&As2[wm][1][lane][0];
    v16bf fb1 = *(const v16bf*)&Bs2[wn][1][lane][0];
    acc = __builtin_amdgcn_wmma_f32_16x16x32_bf16(
        false, fa0, false, fb0, (short)0, acc, false, false);
    acc = __builtin_amdgcn_wmma_f32_16x16x32_bf16(
        false, fa1, false, fb1, (short)0, acc, false, false);
    __syncthreads();
  }

  // epilogue: D VGPR j -> row = hf*8 + j, col = lm
#pragma unroll
  for (int j = 0; j < 8; j++) {
    int gr = m0 + wm * 16 + hf * 8 + j;
    int gc = n0 + wn * 16 + lm;
    if (gr < M && gc < N) {
      float v = acc[j] * alpha;
      if (bias) v += bias[gc];
      if (relu) v = fmaxf(v, 0.f);
      C[(size_t)gr * ldc + gc] = v;
    }
  }
}

// ---------------- gumbel noise + pad + /tau:  la[s,i,j] ----------------
__global__ void k_gumbel(const float* __restrict__ logalpha, float* __restrict__ la, int it) {
  long long idx = (long long)blockIdx.x * blockDim.x + threadIdx.x;
  const long long total = (long long)NSC * PDC * PDC;
  if (idx >= total) return;
  int s = (int)(idx / ((long long)PDC * PDC));
  int rem = (int)(idx % ((long long)PDC * PDC));
  int i = rem / PDC, j = rem % PDC;
  unsigned int h = hash3(0x9E3779B9u * (unsigned)(it + 1) + (unsigned)s,
                         (unsigned)i * 1973u + 0x85ebca6bu,
                         (unsigned)j + 0x27d4eb2fu);
  float u = (float)(h >> 8) * (1.0f / 16777216.0f);
  float g = -__logf(-__logf(u + EPSF) + EPSF);
  float base = (i < N1C && j < N1C) ? logalpha[(size_t)i * N1C + j] : 0.f;
  la[idx] = (base + g) * (1.0f / TAUF);
}

// ---------------- sinkhorn row pass: la -= logsumexp(row) ----------------
__global__ void k_sink_rows(float* __restrict__ la) {
  __shared__ float red[BS];
  float* row = la + (size_t)blockIdx.x * PDC;     // blockIdx.x in [0, NSC*PDC)
  float m = -__builtin_inff();
  for (int j = threadIdx.x; j < PDC; j += blockDim.x) m = fmaxf(m, row[j]);
  red[threadIdx.x] = m; __syncthreads();
  for (int k = BS / 2; k > 0; k >>= 1) {
    if ((int)threadIdx.x < k) red[threadIdx.x] = fmaxf(red[threadIdx.x], red[threadIdx.x + k]);
    __syncthreads();
  }
  m = red[0]; __syncthreads();
  float s = 0.f;
  for (int j = threadIdx.x; j < PDC; j += blockDim.x) s += __expf(row[j] - m);
  red[threadIdx.x] = s; __syncthreads();
  for (int k = BS / 2; k > 0; k >>= 1) {
    if ((int)threadIdx.x < k) red[threadIdx.x] += red[threadIdx.x + k];
    __syncthreads();
  }
  float lse = m + __logf(red[0]);
  for (int j = threadIdx.x; j < PDC; j += blockDim.x) row[j] -= lse;
}

// ---------------- sinkhorn col pass: thread-per-column, online logsumexp ----------------
__global__ void k_sink_cols(float* __restrict__ la) {
  int idx = blockIdx.x * blockDim.x + threadIdx.x;  // over NSC*PDC columns
  if (idx >= NSC * PDC) return;
  int s = idx / PDC, c = idx % PDC;
  float* base = la + (size_t)s * PDC * PDC + c;
  float m = -__builtin_inff(), sum = 0.f;
  for (int i = 0; i < PDC; i++) {
    float x = base[(size_t)i * PDC];
    float nm = fmaxf(m, x);
    sum = sum * __expf(m - nm) + __expf(x - nm);
    m = nm;
  }
  float lse = m + __logf(sum);
  for (int i = 0; i < PDC; i++) base[(size_t)i * PDC] -= lse;
}

// ---------------- reward: W[s,:,e2_0] += theta[s,:,e2_1] ----------------
__global__ void k_rew_scatter(const float* __restrict__ la, const int* __restrict__ e2_0,
                              const int* __restrict__ e2_1, float* __restrict__ W) {
  int e = blockIdx.x, s = blockIdx.y;
  int c0 = e2_0[e], c1 = e2_1[e];
  for (int i = threadIdx.x; i < N1C; i += blockDim.x) {
    float v = __expf(la[((size_t)s * PDC + i) * PDC + c1]);
    atomicAdd(&W[((size_t)s * N1C + i) * N1C + c0], v);
  }
}

// ---------------- reward: matched[s] += sum_j W[s,e1_0,j] * theta[s,e1_1,j] ----------------
__global__ void k_rew_matched(const float* __restrict__ la, const float* __restrict__ W,
                              const int* __restrict__ e1_0, const int* __restrict__ e1_1,
                              float* __restrict__ matched) {
  __shared__ float red[BS];
  int e = blockIdx.x, s = blockIdx.y;
  int a = e1_0[e], b = e1_1[e];
  float sum = 0.f;
  for (int j = threadIdx.x; j < N1C; j += blockDim.x)
    sum += W[((size_t)s * N1C + a) * N1C + j] * __expf(la[((size_t)s * PDC + b) * PDC + j]);
  red[threadIdx.x] = sum; __syncthreads();
  for (int k = BS / 2; k > 0; k >>= 1) {
    if ((int)threadIdx.x < k) red[threadIdx.x] += red[threadIdx.x + k];
    __syncthreads();
  }
  if (threadIdx.x == 0) atomicAdd(&matched[s], red[0]);
}

// ---------------- reward finalize + best tracking (device control flow) ----------------
// scal: [0]=best_reward [1]=loss [2]=loss_count [3]=improved_flag
__global__ void k_finalize(const float* __restrict__ matched, float* __restrict__ scal) {
  if (threadIdx.x != 0) return;
  float racc = 0.f;
  for (int s = 0; s < NSC; s++) {
    float m = matched[s];
    racc += m + MISSV * ((float)NEC - m + ((float)NEC - m));
  }
  float r = (racc / (float)NSC) / (float)NEC;
  if (r > scal[0]) {
    scal[0] = r;
    scal[1] -= r;
    scal[2] += 1.0f;
    scal[3] = 1.0f;
  } else {
    scal[3] = 0.0f;
  }
}

// ---------------- conditional: best_logits <- padded log_alpha; theta_prev <- mean_s theta ----------------
__global__ void k_cond_update(const float* __restrict__ scal, const float* __restrict__ logalpha,
                              const float* __restrict__ la, float* __restrict__ out,
                              float* __restrict__ theta_prev) {
  if (scal[3] == 0.0f) return;
  long long idx = (long long)blockIdx.x * blockDim.x + threadIdx.x;
  if (idx >= (long long)PDC * PDC) return;
  int i = (int)(idx / PDC), j = (int)(idx % PDC);
  float v = (i < N1C && j < N1C) ? logalpha[(size_t)i * N1C + j] : 0.f;
  out[idx] = v;
  if (i < N1C && j < N1C) {
    float acc = 0.f;
    for (int s = 0; s < NSC; s++)
      acc += __expf(la[((size_t)s * PDC + i) * PDC + j]);
    theta_prev[(size_t)i * N1C + j] = acc * (1.0f / (float)NSC);
  }
}

__global__ void k_init_scal(float* __restrict__ scal) {
  if (threadIdx.x == 0) {
    scal[0] = -__builtin_inff();
    scal[1] = 0.f; scal[2] = 0.f; scal[3] = 0.f;
  }
}

__global__ void k_write_loss(const float* __restrict__ scal, float* __restrict__ out) {
  if (threadIdx.x == 0) out[(long long)PDC * PDC] = scal[1] / scal[2];
}

// ============================================================================
extern "C" void kernel_launch(void* const* d_in, const int* in_sizes, int n_in,
                              void* d_out, int out_size, void* d_ws, size_t ws_size,
                              hipStream_t stream) {
  (void)in_sizes; (void)n_in; (void)out_size; (void)ws_size;
  const float* x_s = (const float*)d_in[0];
  const int*   e_s = (const int*)d_in[1];   // [2, NEC]: src row 0, dst row 1
  const float* x_t = (const float*)d_in[2];
  const int*   e_t = (const int*)d_in[3];
  const float* W1  = (const float*)d_in[4];
  const float* b1  = (const float*)d_in[5];
  const float* W2  = (const float*)d_in[6];
  const float* b2  = (const float*)d_in[7];
  float* out = (float*)d_out;

  float* ws = (float*)d_ws;
  size_t off = 0;
  float* la       = ws + off; off += (size_t)NSC * PDC * PDC;  // 32.0M f32
  float* Wrew     = ws + off; off += (size_t)NSC * N1C * N1C;  //  8.0M f32
  float* logalpha = ws + off; off += (size_t)N1C * N1C;
  float* theta    = ws + off; off += (size_t)N1C * N1C;
  float* hbuf     = ws + off; off += (size_t)N1C * DIMC;
  float* agg      = ws + off; off += (size_t)N1C * DIMC;
  float* h1       = ws + off; off += (size_t)N1C * DIMC;
  float* hs       = ws + off; off += (size_t)N1C * DIMC;
  float* ht       = ws + off; off += (size_t)N1C * DIMC;
  float* imp_s    = ws + off; off += N1C;
  float* imp_t    = ws + off; off += N1C;
  float* matched  = ws + off; off += NSC;
  float* scal     = ws + off; off += 8;

  auto g1 = [](long long n) { return dim3((unsigned)((n + BS - 1) / BS)); };

  // init device state (deterministic each call)
  k_init_scal<<<1, 32, 0, stream>>>(scal);
  k_fill<<<g1((long long)N1C * N1C), BS, 0, stream>>>(theta, (long long)N1C * N1C, 0.f);
  k_fill<<<g1((long long)PDC * PDC + 1), BS, 0, stream>>>(out, (long long)PDC * PDC + 1, 0.f);

  const dim3 gemmMLP((DIMC + GTN - 1) / GTN, (N1C + GTM - 1) / GTM);   // 1000x256
  const dim3 gemmSIM((N1C + GTN - 1) / GTN, (N1C + GTM - 1) / GTM);    // 1000x1000
  const dim3 gEdges(NEC, NSC);

  for (int it = 0; it < TITER; it++) {
    // ---- importance ----
    if (it == 0) {
      k_fill<<<g1(N1C), BS, 0, stream>>>(imp_s, N1C, 1.f);
      k_fill<<<g1(N1C), BS, 0, stream>>>(imp_t, N1C, 1.f);
    } else {
      k_row_sum<<<N1C, BS, 0, stream>>>(theta, imp_s, N1C);
      k_col_sum<<<g1(N1C), BS, 0, stream>>>(theta, imp_t, N1C, N1C);
    }

    // ---- f_update for both graphs ----
    const float* xs[2]  = { x_s, x_t };
    const int*   es[2]  = { e_s, e_t };
    const float* imp[2] = { imp_s, imp_t };
    float*       hh[2]  = { hs, ht };
    for (int gix = 0; gix < 2; gix++) {
      k_scale_rows<<<N1C, BS, 0, stream>>>(xs[gix], imp[gix], hbuf, DIMC);
      k_fill<<<g1((long long)N1C * DIMC), BS, 0, stream>>>(agg, (long long)N1C * DIMC, 0.f);
      k_edge_agg<<<NEC, BS, 0, stream>>>(hbuf, es[gix], es[gix] + NEC, agg, DIMC);
      k_gemm_wmma<<<gemmMLP, BS, 0, stream>>>(hbuf, agg, DIMC, W1, DIMC, 0,
                                              b1, 1, 1.0f, h1, DIMC, N1C, DIMC, DIMC);
      k_fill<<<g1((long long)N1C * DIMC), BS, 0, stream>>>(agg, (long long)N1C * DIMC, 0.f);
      k_edge_agg<<<NEC, BS, 0, stream>>>(h1, es[gix], es[gix] + NEC, agg, DIMC);
      k_gemm_wmma<<<gemmMLP, BS, 0, stream>>>(h1, agg, DIMC, W2, DIMC, 0,
                                              b2, 0, 1.0f, hh[gix], DIMC, N1C, DIMC, DIMC);
      k_rownorm<<<N1C, BS, 0, stream>>>(hh[gix], DIMC);
    }

    // ---- similarity: log_alpha = 50 * h_s @ h_t^T ----
    k_gemm_wmma<<<gemmSIM, BS, 0, stream>>>(hs, nullptr, DIMC, ht, DIMC, 1,
                                            nullptr, 0, SIMSCALE, logalpha, N1C,
                                            N1C, N1C, DIMC);

    // ---- gumbel-sinkhorn over padded 8x2000x2000 ----
    k_gumbel<<<g1((long long)NSC * PDC * PDC), BS, 0, stream>>>(logalpha, la, it);
    for (int t = 0; t < SINK_ITERS; t++) {
      k_sink_rows<<<NSC * PDC, BS, 0, stream>>>(la);
      k_sink_cols<<<g1((long long)NSC * PDC), BS, 0, stream>>>(la);
    }

    // ---- reward ----
    k_fill<<<g1((long long)NSC * N1C * N1C), BS, 0, stream>>>(Wrew, (long long)NSC * N1C * N1C, 0.f);
    k_rew_scatter<<<gEdges, BS, 0, stream>>>(la, e_t, e_t + NEC, Wrew);
    k_fill<<<g1(NSC), BS, 0, stream>>>(matched, NSC, 0.f);
    k_rew_matched<<<gEdges, BS, 0, stream>>>(la, Wrew, e_s, e_s + NEC, matched);
    k_finalize<<<1, 32, 0, stream>>>(matched, scal);

    // ---- conditional best update ----
    k_cond_update<<<g1((long long)PDC * PDC), BS, 0, stream>>>(scal, logalpha, la, out, theta);
  }

  k_write_loss<<<1, 32, 0, stream>>>(scal, out);
}
